// GenNet_29738353557560
// MI455X (gfx1250) — matrix operand order repaired and checked by reading, not compiled, not run on hardware
//
#include <hip/hip_runtime.h>
#include <cstdint>
#include <cstddef>

#define NSNP    1000000
#define NGENES  20000
#define NBATCH  64
#define CHUNK   1024
#define GSPAN   128          // max genes a 1024-SNP chunk may span (mean ~21, sigma ~5)
#define BN_EPS  1e-3f

typedef __attribute__((ext_vector_type(4)))  float    f4;
typedef __attribute__((ext_vector_type(4)))  int      i4;
typedef __attribute__((ext_vector_type(8)))  _Float16 h8;
typedef __attribute__((ext_vector_type(16))) _Float16 v16h;
typedef __attribute__((ext_vector_type(8)))  float    v8f;

// Device-scope fp32 atomic add without return (GLOBAL_ATOMIC_ADD_F32, no TH_RETURN).
__device__ __forceinline__ void gfadd(float* p, float v) {
    asm volatile("global_atomic_add_f32 %0, %1, off scope:SCOPE_DEV"
                 :: "v"(p), "v"(v) : "memory");
}

// Native LDS float atomic add (DS_ADD_F32, no return). Generic shared pointers
// carry the LDS offset in their low 32 bits (ISA 10.2).
__device__ __forceinline__ void ldsfadd(float* p, float v) {
    uint32_t off = (uint32_t)(uintptr_t)p;
    asm volatile("ds_add_f32 %0, %1" :: "v"(off), "v"(v) : "memory");
}

// ---------------- Phase 0: zero the gene accumulator in ws ----------------
__global__ __launch_bounds__(256) void k_zero(float* __restrict__ p, int n4) {
    int i = blockIdx.x * 256 + threadIdx.x;
    if (i < n4) { f4 z = {0.f, 0.f, 0.f, 0.f}; ((f4*)p)[i] = z; }
}

// ---------------- Phase A: bandwidth-bound segment scatter-add ----------------
// Block = 1024 contiguous SNPs x all 64 batch rows. Sorted gene_idx => tiny gene
// span; accumulate in LDS [span][64] (pad 65 so concurrent genes hit distinct
// banks), flush boundary-safe with global f32 atomics.
__global__ __launch_bounds__(256) void k_scatter(const float* __restrict__ x,
                                                 const float* __restrict__ w_edge,
                                                 const int*   __restrict__ gidx,
                                                 float*       __restrict__ genes) {
    __shared__ float acc[GSPAN * 65];
    const int tid  = threadIdx.x;
    const int c0   = blockIdx.x * CHUNK;
    const int cend = (c0 + CHUNK < NSNP) ? (c0 + CHUNK) : NSNP;
    const int s    = c0 + tid * 4;                 // NSNP % 4 == 0: f4 all-in or all-out
    const bool active = (s < NSNP);
    const int glo  = gidx[c0];
    const int ghi  = gidx[cend - 1];
    const int span = ghi - glo + 1;

    f4 wv = {0.f, 0.f, 0.f, 0.f};
    i4 gv = {glo, glo, glo, glo};
    if (active) {
        wv = *(const f4*)(w_edge + s);
        gv = *(const i4*)(gidx + s);
    }

    if (span <= GSPAN) {
        for (int i = tid; i < span * 65; i += 256) acc[i] = 0.f;
        __syncthreads();
        #pragma unroll 2
        for (int b = 0; b < NBATCH; ++b) {
            if (active) {
                // stream x non-temporally: 256 MB read once, keep L2 for the accumulator
                f4 xv = __builtin_nontemporal_load((const f4*)(x + (size_t)b * NSNP + s));
                float p0 = xv.x * wv.x, p1 = xv.y * wv.y;
                float p2 = xv.z * wv.z, p3 = xv.w * wv.w;
                if (gv.x == gv.w) {                 // sorted: common case, one gene
                    ldsfadd(&acc[(gv.x - glo) * 65 + b], p0 + p1 + p2 + p3);
                } else {                            // run-merge (<=4 runs)
                    float run = p0; int gp = gv.x;
                    if (gv.y == gp) run += p1; else { ldsfadd(&acc[(gp - glo) * 65 + b], run); gp = gv.y; run = p1; }
                    if (gv.z == gp) run += p2; else { ldsfadd(&acc[(gp - glo) * 65 + b], run); gp = gv.z; run = p2; }
                    if (gv.w == gp) run += p3; else { ldsfadd(&acc[(gp - glo) * 65 + b], run); gp = gv.w; run = p3; }
                    ldsfadd(&acc[(gp - glo) * 65 + b], run);
                }
            }
        }
        // asm ds_add_f32 is outside the compiler's DScnt bookkeeping: drain before barrier
        asm volatile("s_wait_dscnt 0x0" ::: "memory");
        __syncthreads();
        for (int i = tid; i < span * 64; i += 256) {
            const int dg = i >> 6, b = i & 63;
            const float v = acc[dg * 65 + b];
            if (v != 0.f) gfadd(&genes[(size_t)b * NGENES + (glo + dg)], v);
        }
    } else {                                        // pathological span: direct atomics
        if (active) {
            for (int b = 0; b < NBATCH; ++b) {
                f4 xv = __builtin_nontemporal_load((const f4*)(x + (size_t)b * NSNP + s));
                gfadd(&genes[(size_t)b * NGENES + gv.x], xv.x * wv.x);
                gfadd(&genes[(size_t)b * NGENES + gv.y], xv.y * wv.y);
                gfadd(&genes[(size_t)b * NGENES + gv.z], xv.z * wv.z);
                gfadd(&genes[(size_t)b * NGENES + gv.w], xv.w * wv.w);
            }
        }
    }
}

// ---------------- Phase B1: bias + tanh + BN, emit f16 hnorm ----------------
__global__ __launch_bounds__(256) void k_act(const float* __restrict__ genes,
                                             const float* __restrict__ bias,
                                             const float* __restrict__ mean,
                                             const float* __restrict__ var,
                                             _Float16*    __restrict__ h16) {
    const int g = blockIdx.x * 256 + threadIdx.x;
    if (g >= NGENES) return;
    const int b = blockIdx.y;
    float v = genes[(size_t)b * NGENES + g] + bias[g];
    float h = tanhf(v);
    h = (h - mean[g]) * rsqrtf(var[g] + BN_EPS);
    h16[(size_t)b * NGENES + g] = (_Float16)h;
}

// ---------------- Phase B2: 20000-deep dot via v_wmma_f32_16x16x32_f16 ----------
// Block rg handles rows [16rg,16rg+16). Each of 8 waves strides K-chunks of 32
// genes (scalarized wave id -> pure SALU loop control, EXEC stays all-ones for
// WMMA). A = hnorm rows (ISA 16-bit A layout: lane L -> row L%16, K half-select
// by L/16 => two contiguous 16B half loads). B[k][n] = dense_w[k] for all n
// (columns identical -> every D column is the desired dot). f32 accumulation.
__global__ __launch_bounds__(256) void k_head(const _Float16* __restrict__ h16,
                                              const float*    __restrict__ dense_w,
                                              const float*    __restrict__ dense_b,
                                              float*          __restrict__ out) {
    __shared__ float red[8][16];
    const int tid     = threadIdx.x;
    const int w       = tid >> 5;
    const int wS      = __builtin_amdgcn_readfirstlane(w);   // wave-uniform, in SGPR
    const int lane    = tid & 31;
    const int halfSel = lane >> 4;
    const int row     = (blockIdx.x << 4) + (lane & 15);
    const _Float16* rowp = h16 + (size_t)row * NGENES;

    v8f acc = {0.f, 0.f, 0.f, 0.f, 0.f, 0.f, 0.f, 0.f};
    for (int ci = wS; ci < NGENES / 32; ci += 8) {
        const int gbase = ci * 32;
        h8 lo = *(const h8*)(rowp + gbase + 8 * halfSel);        // K = 0..7  (+8 if hi half)
        h8 hi = *(const h8*)(rowp + gbase + 16 + 8 * halfSel);   // K = 16..23(+8)
        v16h a;
        #pragma unroll
        for (int i = 0; i < 8; ++i) { a[i] = lo[i]; a[8 + i] = hi[i]; }
        v16h b;
        const float* wp = dense_w + gbase + 8 * halfSel;         // K = 8*halfSel + i, 16 + 8*halfSel + i
        #pragma unroll
        for (int i = 0; i < 8; ++i) { b[i] = (_Float16)wp[i]; b[8 + i] = (_Float16)wp[16 + i]; }
        acc = __builtin_amdgcn_wmma_f32_16x16x32_f16(
                  false, a, false, b, (short)0, acc, false, false);
    }
    // D layout: vgpr r, lane 0 -> D[M=r][N=0]; lane 16 -> D[M=8+r][N=0]
    if (lane == 0) {
        #pragma unroll
        for (int r = 0; r < 8; ++r) red[w][r] = acc[r];
    }
    if (lane == 16) {
        #pragma unroll
        for (int r = 0; r < 8; ++r) red[w][8 + r] = acc[r];
    }
    __syncthreads();
    if (tid < 16) {
        float s = 0.f;
        #pragma unroll
        for (int ww = 0; ww < 8; ++ww) s += red[ww][tid];
        s += dense_b[0];
        out[(blockIdx.x << 4) + tid] = 1.f / (1.f + expf(-s));
    }
}

extern "C" void kernel_launch(void* const* d_in, const int* in_sizes, int n_in,
                              void* d_out, int out_size, void* d_ws, size_t ws_size,
                              hipStream_t stream) {
    (void)in_sizes; (void)n_in; (void)out_size; (void)ws_size;
    const float* x          = (const float*)d_in[0];
    const float* w_edge     = (const float*)d_in[1];
    const float* gene_bias  = (const float*)d_in[2];
    const float* bn_mean    = (const float*)d_in[3];
    const float* bn_var     = (const float*)d_in[4];
    const float* dense_w    = (const float*)d_in[5];
    const float* dense_b    = (const float*)d_in[6];
    const int*   gene_idx   = (const int*)d_in[7];
    float* outp = (float*)d_out;

    // ws layout: [0, 5.12MB) f32 genes accumulator; [5.12MB, 7.68MB) f16 hnorm
    float*    genes = (float*)d_ws;
    _Float16* h16   = (_Float16*)((char*)d_ws + (size_t)NBATCH * NGENES * sizeof(float));

    const int n4 = NBATCH * NGENES / 4;                       // 320000
    k_zero<<<(n4 + 255) / 256, 256, 0, stream>>>(genes, n4);  // 1250 blocks
    k_scatter<<<(NSNP + CHUNK - 1) / CHUNK, 256, 0, stream>>>(x, w_edge, gene_idx, genes);
    k_act<<<dim3((NGENES + 255) / 256, NBATCH), 256, 0, stream>>>(genes, gene_bias,
                                                                  bn_mean, bn_var, h16);
    k_head<<<NBATCH / 16, 256, 0, stream>>>(h16, dense_w, dense_b, outp);
}
